// FirePredictorGAT_54692113547828
// MI455X (gfx1250) — matrix operand order repaired and checked by reading, not compiled, not run on hardware
//
#include <hip/hip_runtime.h>
#include <hip/hip_bf16.h>

typedef __attribute__((ext_vector_type(16))) _Float16 v16h;
typedef __attribute__((ext_vector_type(8)))  float    v8f;

#define FIN   128
#define FOUT  256   // heads(8) * ch(32)
#define HEADS 8
#define CH    32
#define NEG_SLOPE 0.2f

// ---------------------------------------------------------------- utilities
__global__ void fill_f32_kernel(float* __restrict__ p, float v, int n) {
    int t = blockIdx.x * blockDim.x + threadIdx.x;
    if (t < n) p[t] = v;
}

__device__ __forceinline__ void atomicMaxFloat(float* addr, float val) {
    // ordered-int trick; addr initialized to -inf
    if (val >= 0.0f) atomicMax((int*)addr, __float_as_int(val));
    else             atomicMin((unsigned int*)addr, __float_as_uint(val));
}

__device__ __forceinline__ void edge_sd(const int* __restrict__ ei, int E_raw,
                                        int e, int& s, int& d) {
    if (e < E_raw) { s = ei[e]; d = ei[E_raw + e]; }
    else           { s = e - E_raw; d = s; }          // self loops appended
}

// W1t[n,k] = W1[k,n]; 128x256 -> 256x128 (one-time, 128 KB)
__global__ void transpose_w1_kernel(const float* __restrict__ W,
                                    float* __restrict__ Wt) {
    int t = blockIdx.x * blockDim.x + threadIdx.x;
    if (t >= FIN * FOUT) return;
    const int k = t >> 8;          // / FOUT
    const int n = t & (FOUT - 1);
    Wt[(size_t)n * FIN + k] = W[t];
}

// ---------------------------------------------------------------- GEMM1 (WMMA)
// H[M,256] = X[M,128] * W1[128,256]; one 16x16 tile per wave; K unrolled 4x32.
// B operands come from W1t (row-major [n,k]) so all loads are float4.
#define PACK4(dst, base, f4)                      \
    dst[(base) + 0] = (_Float16)(f4).x;           \
    dst[(base) + 1] = (_Float16)(f4).y;           \
    dst[(base) + 2] = (_Float16)(f4).z;           \
    dst[(base) + 3] = (_Float16)(f4).w;

__global__ __launch_bounds__(256)
void gemm1_wmma_kernel(const float* __restrict__ X, const float* __restrict__ Wt,
                       float* __restrict__ H, int M) {
    const int lane = threadIdx.x & 31;
    const int wave = threadIdx.x >> 5;
    const int tile = blockIdx.x * 8 + wave;
    const int mt = tile >> 4;            // FOUT/16 == 16 N-tiles
    const int nt = tile & 15;
    if (mt * 16 >= M) return;            // uniform per wave (never hits for M=50000)
    const int m0 = mt * 16, n0 = nt * 16;
    const int r  = lane & 15;            // A row / B,D column
    const int hi = lane >> 4;

    const float* __restrict__ arow = X  + (size_t)(m0 + r) * FIN;
    const float* __restrict__ brow = Wt + (size_t)(n0 + r) * FIN;

    v8f acc = {};
#pragma unroll
    for (int kk = 0; kk < FIN; kk += 32) {
        // A 16x32 f16 layout: lane-half hi selects K runs [kk+8hi..+7] and [kk+16+8hi..+7]
        const float4 a0 = *(const float4*)(arow + kk + 8 * hi);
        const float4 a1 = *(const float4*)(arow + kk + 8 * hi + 4);
        const float4 a2 = *(const float4*)(arow + kk + 16 + 8 * hi);
        const float4 a3 = *(const float4*)(arow + kk + 16 + 8 * hi + 4);
        // B 32x16 f16 layout: lane-half hi selects K runs [kk+8hi..+7] and [kk+16+8hi..+7]
        const float4 b0 = *(const float4*)(brow + kk + 8 * hi);
        const float4 b1 = *(const float4*)(brow + kk + 8 * hi + 4);
        const float4 b2 = *(const float4*)(brow + kk + 16 + 8 * hi);
        const float4 b3 = *(const float4*)(brow + kk + 16 + 8 * hi + 4);
        v16h a, b;
        PACK4(a, 0, a0)  PACK4(a, 4, a1)  PACK4(a, 8, a2)  PACK4(a, 12, a3)
        PACK4(b, 0, b0)  PACK4(b, 4, b1)  PACK4(b, 8, b2)  PACK4(b, 12, b3)
        acc = __builtin_amdgcn_wmma_f32_16x16x32_f16(
                  false, a, false, b, (short)0, acc, false, false);
    }
#pragma unroll
    for (int v = 0; v < 8; ++v)
        H[(size_t)(m0 + v + 8 * hi) * FOUT + n0 + r] = acc[v];
}

// ---------------------------------------------------------------- attention coefs
__global__ void attn_coef_kernel(const float* __restrict__ H,
                                 const float* __restrict__ att_s,
                                 const float* __restrict__ att_d,
                                 float* __restrict__ as_, float* __restrict__ ad_,
                                 int N) {
    int t = blockIdx.x * blockDim.x + threadIdx.x;      // (n,h)
    if (t >= N * HEADS) return;
    const int n = t >> 3, h = t & 7;
    const float4* hv = (const float4*)(H + (size_t)n * FOUT + h * CH);
    const float4* sv = (const float4*)(att_s + h * CH);
    const float4* dv = (const float4*)(att_d + h * CH);
    float ss = 0.0f, dd = 0.0f;
#pragma unroll
    for (int i = 0; i < CH / 4; ++i) {
        float4 hh = hv[i], s4 = sv[i], d4 = dv[i];
        ss += hh.x * s4.x + hh.y * s4.y + hh.z * s4.z + hh.w * s4.w;
        dd += hh.x * d4.x + hh.y * d4.y + hh.z * d4.z + hh.w * d4.w;
    }
    as_[t] = ss; ad_[t] = dd;
}

// ---------------------------------------------------------------- L1 edge passes
__global__ void edge_max1_kernel(const int* __restrict__ ei, int E_raw, int E_tot,
                                 const float* __restrict__ as_,
                                 const float* __restrict__ ad_,
                                 float* __restrict__ mx) {
    int t = blockIdx.x * blockDim.x + threadIdx.x;
    if (t >= E_tot * HEADS) return;
    int e = t >> 3, h = t & 7, s, d;
    edge_sd(ei, E_raw, e, s, d);
    float v = as_[s * HEADS + h] + ad_[d * HEADS + h];
    v = (v >= 0.0f) ? v : NEG_SLOPE * v;
    atomicMaxFloat(&mx[d * HEADS + h], v);
}

__global__ void edge_sum1_kernel(const int* __restrict__ ei, int E_raw, int E_tot,
                                 const float* __restrict__ as_,
                                 const float* __restrict__ ad_,
                                 const float* __restrict__ mx,
                                 float* __restrict__ den) {
    int t = blockIdx.x * blockDim.x + threadIdx.x;
    if (t >= E_tot * HEADS) return;
    int e = t >> 3, h = t & 7, s, d;
    edge_sd(ei, E_raw, e, s, d);
    float v = as_[s * HEADS + h] + ad_[d * HEADS + h];
    v = (v >= 0.0f) ? v : NEG_SLOPE * v;
    atomicAdd(&den[d * HEADS + h], __expf(v - mx[d * HEADS + h]));
}

__global__ void edge_agg1_kernel(const int* __restrict__ ei, int E_raw, int total,
                                 const float* __restrict__ as_,
                                 const float* __restrict__ ad_,
                                 const float* __restrict__ mx,
                                 const float* __restrict__ den,
                                 const float* __restrict__ H,
                                 float* __restrict__ out) {
    int t = blockIdx.x * blockDim.x + threadIdx.x;      // (edge, 4-ch group)
    if (t >= total) return;
    const int e  = t >> 6;
    const int ch = (t & 63) * 4;
    const int h  = ch >> 5;
    int s, d; edge_sd(ei, E_raw, e, s, d);
    float v = as_[s * HEADS + h] + ad_[d * HEADS + h];
    v = (v >= 0.0f) ? v : NEG_SLOPE * v;
    const float alpha = __expf(v - mx[d * HEADS + h]) / den[d * HEADS + h];
    const float4 hv = *(const float4*)(H + (size_t)s * FOUT + ch);
    float* ob = out + (size_t)d * FOUT + ch;
    atomicAdd(ob + 0, alpha * hv.x);
    atomicAdd(ob + 1, alpha * hv.y);
    atomicAdd(ob + 2, alpha * hv.z);
    atomicAdd(ob + 3, alpha * hv.w);
}

__global__ void bias_elu_kernel(float* __restrict__ o1,
                                const float* __restrict__ b1, int total) {
    int t = blockIdx.x * blockDim.x + threadIdx.x;
    if (t >= total) return;
    float v = o1[t] + b1[t & (FOUT - 1)];
    o1[t] = (v > 0.0f) ? v : (__expf(v) - 1.0f);        // ELU
}

// ---------------------------------------------------------------- layer 2
__global__ __launch_bounds__(256)
void matvec2_kernel(const float* __restrict__ Hl, const float* __restrict__ W2,
                    float* __restrict__ h2, int N) {
    const int lane = threadIdx.x & 31;
    const int node = (blockIdx.x * blockDim.x + threadIdx.x) >> 5;   // wave per node
    if (node >= N) return;
    float sum = 0.0f;
#pragma unroll
    for (int j = 0; j < FOUT / 32; ++j) {
        const int k = lane + 32 * j;
        sum += Hl[(size_t)node * FOUT + k] * W2[k];
    }
#pragma unroll
    for (int off = 16; off > 0; off >>= 1) sum += __shfl_xor(sum, off, 32);
    if (lane == 0) h2[node] = sum;
}

__global__ void edge_max2_kernel(const int* __restrict__ ei, int E_raw, int E_tot,
                                 const float* __restrict__ h2,
                                 const float* __restrict__ aS,
                                 const float* __restrict__ aD,
                                 float* __restrict__ mx) {
    int e = blockIdx.x * blockDim.x + threadIdx.x;
    if (e >= E_tot) return;
    int s, d; edge_sd(ei, E_raw, e, s, d);
    float v = h2[s] * aS[0] + h2[d] * aD[0];
    v = (v >= 0.0f) ? v : NEG_SLOPE * v;
    atomicMaxFloat(&mx[d], v);
}

__global__ void edge_sum2_kernel(const int* __restrict__ ei, int E_raw, int E_tot,
                                 const float* __restrict__ h2,
                                 const float* __restrict__ aS,
                                 const float* __restrict__ aD,
                                 const float* __restrict__ mx,
                                 float* __restrict__ den) {
    int e = blockIdx.x * blockDim.x + threadIdx.x;
    if (e >= E_tot) return;
    int s, d; edge_sd(ei, E_raw, e, s, d);
    float v = h2[s] * aS[0] + h2[d] * aD[0];
    v = (v >= 0.0f) ? v : NEG_SLOPE * v;
    atomicAdd(&den[d], __expf(v - mx[d]));
}

__global__ void edge_agg2_kernel(const int* __restrict__ ei, int E_raw, int E_tot,
                                 const float* __restrict__ h2,
                                 const float* __restrict__ aS,
                                 const float* __restrict__ aD,
                                 const float* __restrict__ mx,
                                 const float* __restrict__ den,
                                 float* __restrict__ out2) {
    int e = blockIdx.x * blockDim.x + threadIdx.x;
    if (e >= E_tot) return;
    int s, d; edge_sd(ei, E_raw, e, s, d);
    float v = h2[s] * aS[0] + h2[d] * aD[0];
    v = (v >= 0.0f) ? v : NEG_SLOPE * v;
    const float alpha = __expf(v - mx[d]) / den[d];
    atomicAdd(&out2[d], alpha * h2[s]);
}

__global__ void finalize_kernel(const float* __restrict__ out2,
                                const float* __restrict__ b2,
                                float* __restrict__ out, int N) {
    int t = blockIdx.x * blockDim.x + threadIdx.x;
    if (t < N) out[t] = out2[t] + b2[0];   // heads=1 -> mean == identity
}

// ---------------------------------------------------------------- launcher
extern "C" void kernel_launch(void* const* d_in, const int* in_sizes, int n_in,
                              void* d_out, int out_size, void* d_ws, size_t ws_size,
                              hipStream_t stream) {
    const float* x     = (const float*)d_in[0];
    const int*   ei    = (const int*)  d_in[1];
    const float* W1    = (const float*)d_in[2];
    const float* attS1 = (const float*)d_in[3];
    const float* attD1 = (const float*)d_in[4];
    const float* b1    = (const float*)d_in[5];
    const float* W2    = (const float*)d_in[6];
    const float* attS2 = (const float*)d_in[7];
    const float* attD2 = (const float*)d_in[8];
    const float* b2    = (const float*)d_in[9];
    float* out = (float*)d_out;

    const int N     = in_sizes[0] / FIN;     // 50000
    const int E_raw = in_sizes[1] / 2;       // 800000
    const int E_tot = E_raw + N;             // + self loops

    // workspace carve-up (floats)
    float* ws = (float*)d_ws;
    size_t o = 0;
    float* h1   = ws + o;  o += (size_t)N * FOUT;   // 12.8M
    float* o1   = ws + o;  o += (size_t)N * FOUT;   // 12.8M
    float* as1  = ws + o;  o += (size_t)N * HEADS;
    float* ad1  = ws + o;  o += (size_t)N * HEADS;
    float* mx1  = ws + o;  o += (size_t)N * HEADS;
    float* den1 = ws + o;  o += (size_t)N * HEADS;
    float* h2   = ws + o;  o += (size_t)N;
    float* mx2  = ws + o;  o += (size_t)N;
    float* den2 = ws + o;  o += (size_t)N;
    float* out2 = ws + o;  o += (size_t)N;
    float* W1t  = ws + o;  o += (size_t)FIN * FOUT; // 32768

    const int TB = 256;
    const float NEG_INF = -__builtin_huge_valf();
    auto blocks = [](long long n, int tb) { return (int)((n + tb - 1) / tb); };

    // init accumulators (ws is poisoned by harness)
    fill_f32_kernel<<<blocks((long long)N * FOUT, TB), TB, 0, stream>>>(o1, 0.0f, N * FOUT);
    fill_f32_kernel<<<blocks(N * HEADS, TB), TB, 0, stream>>>(mx1, NEG_INF, N * HEADS);
    fill_f32_kernel<<<blocks(N * HEADS, TB), TB, 0, stream>>>(den1, 0.0f, N * HEADS);
    fill_f32_kernel<<<blocks(N, TB), TB, 0, stream>>>(mx2, NEG_INF, N);
    fill_f32_kernel<<<blocks(N, TB), TB, 0, stream>>>(den2, 0.0f, N);
    fill_f32_kernel<<<blocks(N, TB), TB, 0, stream>>>(out2, 0.0f, N);

    // layer 1
    transpose_w1_kernel<<<blocks(FIN * FOUT, TB), TB, 0, stream>>>(W1, W1t);
    const int mTiles = (N + 15) / 16;                 // 3125
    const int tiles  = mTiles * (FOUT / 16);          // 50000
    gemm1_wmma_kernel<<<(tiles + 7) / 8, 256, 0, stream>>>(x, W1t, h1, N);

    attn_coef_kernel<<<blocks(N * HEADS, TB), TB, 0, stream>>>(h1, attS1, attD1, as1, ad1, N);
    edge_max1_kernel<<<blocks((long long)E_tot * HEADS, TB), TB, 0, stream>>>(
        ei, E_raw, E_tot, as1, ad1, mx1);
    edge_sum1_kernel<<<blocks((long long)E_tot * HEADS, TB), TB, 0, stream>>>(
        ei, E_raw, E_tot, as1, ad1, mx1, den1);
    const int aggTotal = E_tot * 64;                  // 4 channels / thread
    edge_agg1_kernel<<<blocks(aggTotal, TB), TB, 0, stream>>>(
        ei, E_raw, aggTotal, as1, ad1, mx1, den1, h1, o1);
    bias_elu_kernel<<<blocks((long long)N * FOUT, TB), TB, 0, stream>>>(o1, b1, N * FOUT);

    // layer 2
    matvec2_kernel<<<blocks((long long)N * 32, 256), 256, 0, stream>>>(o1, W2, h2, N);
    edge_max2_kernel<<<blocks(E_tot, TB), TB, 0, stream>>>(ei, E_raw, E_tot, h2, attS2, attD2, mx2);
    edge_sum2_kernel<<<blocks(E_tot, TB), TB, 0, stream>>>(ei, E_raw, E_tot, h2, attS2, attD2, mx2, den2);
    edge_agg2_kernel<<<blocks(E_tot, TB), TB, 0, stream>>>(ei, E_raw, E_tot, h2, attS2, attD2, mx2, den2, out2);
    finalize_kernel<<<blocks(N, TB), TB, 0, stream>>>(out2, b2, out, N);
}
